// GNBlock_19851338842524
// MI455X (gfx1250) — compile-verified, matrix operand order checked by reading, not run on hardware
//
#include <hip/hip_runtime.h>

// ---------------------------------------------------------------------------
// CDNA5 (gfx1250) GNN MetaLayer block: edge MLP + scatter-add + node MLP +
// global MLP. Matmuls run on v_wmma_f32_16x16x32_bf16 (wave32, 16x16 tiles,
// f32 accumulate). LayerNorm/ReLU fused in registers via half-wave shuffles.
// ---------------------------------------------------------------------------

typedef __attribute__((ext_vector_type(16))) __bf16         v16bf;
typedef __attribute__((ext_vector_type(16))) unsigned short v16u;
typedef __attribute__((ext_vector_type(8)))  unsigned short v8u;
typedef __attribute__((ext_vector_type(8)))  float          v8f;

#define LN_EPS 1e-5f

__device__ __forceinline__ unsigned short f2bf(float f) {
  unsigned int u = __builtin_bit_cast(unsigned int, f);
  unsigned int r = 0x7FFFu + ((u >> 16) & 1u);
  return (unsigned short)((u + r) >> 16);
}

// A-fragment (16x32 bf16) from a global fp32 row: per documented layout,
// lane needs row[kb..kb+7] and row[kb+16..kb+23] (kb = k0 + 8*half).
__device__ __forceinline__ v16bf wfrag_global(const float* p) {
  const float4* q = (const float4*)p;
  float4 a0 = q[0], a1 = q[1], a2 = q[4], a3 = q[5];
  v16u u;
  u[0]=f2bf(a0.x);  u[1]=f2bf(a0.y);  u[2]=f2bf(a0.z);  u[3]=f2bf(a0.w);
  u[4]=f2bf(a1.x);  u[5]=f2bf(a1.y);  u[6]=f2bf(a1.z);  u[7]=f2bf(a1.w);
  u[8]=f2bf(a2.x);  u[9]=f2bf(a2.y);  u[10]=f2bf(a2.z); u[11]=f2bf(a2.w);
  u[12]=f2bf(a3.x); u[13]=f2bf(a3.y); u[14]=f2bf(a3.z); u[15]=f2bf(a3.w);
  return __builtin_bit_cast(v16bf, u);
}

// Fragment from LDS bf16 (two contiguous 16B chunks at kb and kb+16).
__device__ __forceinline__ v16bf wfrag_lds(const unsigned short* p) {
  v8u lo = *(const v8u*)p;
  v8u hi = *(const v8u*)(p + 16);
  v16u u;
#pragma unroll
  for (int i = 0; i < 8; ++i) { u[i] = lo[i]; u[8 + i] = hi[i]; }
  return __builtin_bit_cast(v16bf, u);
}

__device__ __forceinline__ v8f wmma_bf16(v16bf a, v16bf b, v8f c) {
  return __builtin_amdgcn_wmma_f32_16x16x32_bf16(false, a, false, b,
                                                 (short)0, c, false, false);
}

// Sum over the 16 lanes of a half-wave (xor masks 1..8 stay within halves).
__device__ __forceinline__ float half16_sum(float v) {
  v += __shfl_xor(v, 1, 32);
  v += __shfl_xor(v, 2, 32);
  v += __shfl_xor(v, 4, 32);
  v += __shfl_xor(v, 8, 32);
  return v;
}

// ReLU + row LayerNorm, fully in registers. C layout: VGPR r holds row r
// (lanes 0-15) / row r+8 (lanes 16-31); each lane covers one column per tile.
__device__ __forceinline__ void relu_ln_rows(v8f (&acc)[8], const float* gv,
                                             const float* bev) {
#pragma unroll
  for (int t = 0; t < 8; ++t)
#pragma unroll
    for (int r = 0; r < 8; ++r) acc[t][r] = fmaxf(acc[t][r], 0.0f);
#pragma unroll
  for (int r = 0; r < 8; ++r) {
    float s = 0.f, ss = 0.f;
#pragma unroll
    for (int t = 0; t < 8; ++t) { float v = acc[t][r]; s += v; ss += v * v; }
    s = half16_sum(s); ss = half16_sum(ss);
    float m    = s  * (1.0f / 128.0f);
    float var  = ss * (1.0f / 128.0f) - m * m;
    float rinv = rsqrtf(var + LN_EPS);
#pragma unroll
    for (int t = 0; t < 8; ++t)
      acc[t][r] = (acc[t][r] - m) * rinv * gv[t] + bev[t];
  }
}

// Store a wave's 16x128 activations to LDS as bf16 (row-major) so the next
// layer can read A-fragments as contiguous 16B chunks.
__device__ __forceinline__ void store_h(unsigned short* hb, const v8f (&acc)[8],
                                        int wv, int ln, int hf) {
#pragma unroll
  for (int t = 0; t < 8; ++t)
#pragma unroll
    for (int r = 0; r < 8; ++r)
      hb[(wv * 16 + r + hf * 8) * 128 + t * 16 + ln] = f2bf(acc[t][r]);
}

// Dense 128-wide layer: activations (LDS bf16, row-major) x W^T (LDS bf16).
__device__ __forceinline__ void layer_lds(v8f (&acc)[8],
                                          const unsigned short* arow,
                                          const unsigned short* wt,
                                          const float* bv, int ln, int hf) {
#pragma unroll
  for (int t = 0; t < 8; ++t) {
    v8f c;
#pragma unroll
    for (int r = 0; r < 8; ++r) c[r] = bv[t];
    acc[t] = c;
  }
#pragma unroll
  for (int k0 = 0; k0 < 128; k0 += 32) {
    v16bf af = wfrag_lds(arow + k0 + hf * 8);
#pragma unroll
    for (int t = 0; t < 8; ++t) {
      v16bf bf = wfrag_lds(wt + (t * 16 + ln) * 128 + k0 + hf * 8);
      acc[t] = wmma_bf16(af, bf, acc[t]);
    }
  }
}

// ---------------------------------------------------------------------------
// Edge MLP: 128 edges per block, 8 waves x (16 rows x 128 cols).
// Fuses gather, 3-layer MLP, edge output store, scatter-add and edge_sum.
// ---------------------------------------------------------------------------
__global__ __launch_bounds__(256) void edge_mlp_kernel(
    const float* __restrict__ x, const int* __restrict__ esrc,
    const int* __restrict__ edst, const float* __restrict__ eattr,
    const float* __restrict__ u,
    const float* __restrict__ W1, const float* __restrict__ b1,
    const float* __restrict__ g1, const float* __restrict__ be1,
    const float* __restrict__ W2, const float* __restrict__ b2,
    const float* __restrict__ g2, const float* __restrict__ be2,
    const float* __restrict__ W3, const float* __restrict__ b3,
    float* __restrict__ eout, float* __restrict__ agg,
    float* __restrict__ edge_sum) {
  extern __shared__ unsigned short smem_us[];
  unsigned short* w1t = smem_us;                  // phase A: [128][512]
  unsigned short* w2t = smem_us;                  // phase B overlays w1t:
  unsigned short* w3t = smem_us + 128 * 128;
  unsigned short* h1b = smem_us + 2 * 128 * 128;
  unsigned short* h2b = smem_us + 3 * 128 * 128;

  const int tid  = threadIdx.x;
  const int lane = tid & 31, wv = tid >> 5;
  const int ln   = lane & 15, hf = lane >> 4;
  const int rowbase = blockIdx.x * 128 + wv * 16;

  // Stage W1^T as bf16 (coalesced global reads; scattered LDS writes).
  for (int idx = tid; idx < 512 * 128; idx += 256) {
    int k = idx >> 7, n = idx & 127;
    w1t[n * 512 + k] = f2bf(W1[idx]);
  }
  __syncthreads();

  const int e = rowbase + ln;
  const float* p0 = x + (long)esrc[e] * 128;   // src features
  const float* p1 = x + (long)edst[e] * 128;   // dst features
  const float* p2 = eattr + (long)e * 128;     // edge features
  const float* p3 = u;                         // globals

  float bv[8], gv[8], bev[8];
#pragma unroll
  for (int t = 0; t < 8; ++t) {
    int n = t * 16 + ln;
    bv[t] = b1[n]; gv[t] = g1[n]; bev[t] = be1[n];
  }

  v8f acc[8];
#pragma unroll
  for (int t = 0; t < 8; ++t) {
    v8f c;
#pragma unroll
    for (int r = 0; r < 8; ++r) c[r] = bv[t];
    acc[t] = c;
  }

  // Layer 1: K = 512 (4 concat segments of 128), 16 k-steps x 8 n-tiles.
#pragma unroll
  for (int k0 = 0; k0 < 512; k0 += 32) {
    const float* base = (k0 < 128) ? p0 : (k0 < 256) ? p1 : (k0 < 384) ? p2 : p3;
    v16bf af = wfrag_global(base + (k0 & 127) + hf * 8);
#pragma unroll
    for (int t = 0; t < 8; ++t) {
      v16bf bf = wfrag_lds(w1t + (t * 16 + ln) * 512 + k0 + hf * 8);
      acc[t] = wmma_bf16(af, bf, acc[t]);
    }
  }
  relu_ln_rows(acc, gv, bev);
  __syncthreads();  // all waves done with w1t region

  // Stage W2^T, W3^T into the reclaimed region; spill h1 as bf16.
  for (int idx = tid; idx < 128 * 128; idx += 256) {
    int k = idx >> 7, n = idx & 127;
    w2t[n * 128 + k] = f2bf(W2[idx]);
    w3t[n * 128 + k] = f2bf(W3[idx]);
  }
  store_h(h1b, acc, wv, ln, hf);
  __syncthreads();

  // Layer 2
#pragma unroll
  for (int t = 0; t < 8; ++t) {
    int n = t * 16 + ln;
    bv[t] = b2[n]; gv[t] = g2[n]; bev[t] = be2[n];
  }
  layer_lds(acc, h1b + (wv * 16 + ln) * 128, w2t, bv, ln, hf);
  relu_ln_rows(acc, gv, bev);
  store_h(h2b, acc, wv, ln, hf);
  __syncthreads();

  // Layer 3 (no activation)
#pragma unroll
  for (int t = 0; t < 8; ++t) bv[t] = b3[t * 16 + ln];
  layer_lds(acc, h2b + (wv * 16 + ln) * 128, w3t, bv, ln, hf);

  // Store edge output, fused scatter-add to agg[dst], and edge_sum partials.
  int dstm[8];
#pragma unroll
  for (int r = 0; r < 8; ++r) dstm[r] = edst[rowbase + r + hf * 8];
#pragma unroll
  for (int t = 0; t < 8; ++t) {
    int n = t * 16 + ln;
    float cs = 0.f;
#pragma unroll
    for (int r = 0; r < 8; ++r) {
      float v = acc[t][r];
      eout[(long)(rowbase + r + hf * 8) * 128 + n] = v;
      atomicAdd(&agg[(long)dstm[r] * 128 + n], v);
      cs += v;
    }
    cs += __shfl_xor(cs, 16, 32);
    if (hf == 0) atomicAdd(&edge_sum[n], cs);
  }
}

// ---------------------------------------------------------------------------
// Node MLP: input concat [x | agg | u] (K = 384), same WMMA pipeline.
// ---------------------------------------------------------------------------
__global__ __launch_bounds__(256) void node_mlp_kernel(
    const float* __restrict__ x, const float* __restrict__ agg,
    const float* __restrict__ u,
    const float* __restrict__ W1, const float* __restrict__ b1,
    const float* __restrict__ g1, const float* __restrict__ be1,
    const float* __restrict__ W2, const float* __restrict__ b2,
    const float* __restrict__ g2, const float* __restrict__ be2,
    const float* __restrict__ W3, const float* __restrict__ b3,
    float* __restrict__ xout, float* __restrict__ node_sum, int N) {
  extern __shared__ unsigned short smem_us[];
  unsigned short* w1t = smem_us;                  // phase A: [128][384]
  unsigned short* w2t = smem_us;
  unsigned short* w3t = smem_us + 128 * 128;
  unsigned short* h1b = smem_us + 2 * 128 * 128;
  unsigned short* h2b = smem_us + 3 * 128 * 128;

  const int tid  = threadIdx.x;
  const int lane = tid & 31, wv = tid >> 5;
  const int ln   = lane & 15, hf = lane >> 4;
  const int rowbase = blockIdx.x * 128 + wv * 16;

  for (int idx = tid; idx < 384 * 128; idx += 256) {
    int k = idx >> 7, n = idx & 127;
    w1t[n * 384 + k] = f2bf(W1[idx]);
  }
  __syncthreads();

  int arow = rowbase + ln;
  if (arow > N - 1) arow = N - 1;  // clamp loads; stores guarded below
  const float* p0 = x + (long)arow * 128;
  const float* p1 = agg + (long)arow * 128;
  const float* p2 = u;

  float bv[8], gv[8], bev[8];
#pragma unroll
  for (int t = 0; t < 8; ++t) {
    int n = t * 16 + ln;
    bv[t] = b1[n]; gv[t] = g1[n]; bev[t] = be1[n];
  }

  v8f acc[8];
#pragma unroll
  for (int t = 0; t < 8; ++t) {
    v8f c;
#pragma unroll
    for (int r = 0; r < 8; ++r) c[r] = bv[t];
    acc[t] = c;
  }

#pragma unroll
  for (int k0 = 0; k0 < 384; k0 += 32) {
    const float* base = (k0 < 128) ? p0 : (k0 < 256) ? p1 : p2;
    v16bf af = wfrag_global(base + (k0 & 127) + hf * 8);
#pragma unroll
    for (int t = 0; t < 8; ++t) {
      v16bf bf = wfrag_lds(w1t + (t * 16 + ln) * 384 + k0 + hf * 8);
      acc[t] = wmma_bf16(af, bf, acc[t]);
    }
  }
  relu_ln_rows(acc, gv, bev);
  __syncthreads();

  for (int idx = tid; idx < 128 * 128; idx += 256) {
    int k = idx >> 7, n = idx & 127;
    w2t[n * 128 + k] = f2bf(W2[idx]);
    w3t[n * 128 + k] = f2bf(W3[idx]);
  }
  store_h(h1b, acc, wv, ln, hf);
  __syncthreads();

#pragma unroll
  for (int t = 0; t < 8; ++t) {
    int n = t * 16 + ln;
    bv[t] = b2[n]; gv[t] = g2[n]; bev[t] = be2[n];
  }
  layer_lds(acc, h1b + (wv * 16 + ln) * 128, w2t, bv, ln, hf);
  relu_ln_rows(acc, gv, bev);
  store_h(h2b, acc, wv, ln, hf);
  __syncthreads();

#pragma unroll
  for (int t = 0; t < 8; ++t) bv[t] = b3[t * 16 + ln];
  layer_lds(acc, h2b + (wv * 16 + ln) * 128, w3t, bv, ln, hf);

#pragma unroll
  for (int t = 0; t < 8; ++t) {
    int n = t * 16 + ln;
    float cs = 0.f;
#pragma unroll
    for (int r = 0; r < 8; ++r) {
      int m = rowbase + r + hf * 8;
      float v = acc[t][r];
      if (m < N) { xout[(long)m * 128 + n] = v; cs += v; }
    }
    cs += __shfl_xor(cs, 16, 32);
    if (hf == 0) atomicAdd(&node_sum[n], cs);
  }
}

// ---------------------------------------------------------------------------
// Global MLP: [1,384] -> 128 -> 128 -> 1. Tiny; plain VALU, one block.
// ---------------------------------------------------------------------------
__global__ __launch_bounds__(128) void glob_mlp_kernel(
    const float* __restrict__ u, const float* __restrict__ node_sum,
    const float* __restrict__ edge_sum,
    const float* __restrict__ W1, const float* __restrict__ b1,
    const float* __restrict__ g1, const float* __restrict__ be1,
    const float* __restrict__ W2, const float* __restrict__ b2,
    const float* __restrict__ g2, const float* __restrict__ be2,
    const float* __restrict__ W3, const float* __restrict__ b3,
    float* __restrict__ out) {
  __shared__ float fin[384];
  __shared__ float buf[128];
  __shared__ float red[2];
  const int t = threadIdx.x;
  fin[t] = u[t]; fin[128 + t] = node_sum[t]; fin[256 + t] = edge_sum[t];
  __syncthreads();

  float a = b1[t];
  for (int i = 0; i < 384; ++i) a = fmaf(fin[i], W1[i * 128 + t], a);
  a = fmaxf(a, 0.f);
  buf[t] = a; __syncthreads();
  if (t == 0) { float s = 0; for (int i = 0; i < 128; ++i) s += buf[i]; red[0] = s * (1.f/128.f); }
  __syncthreads();
  float d = a - red[0];
  buf[t] = d * d; __syncthreads();
  if (t == 0) { float s = 0; for (int i = 0; i < 128; ++i) s += buf[i]; red[1] = s * (1.f/128.f); }
  __syncthreads();
  a = d * rsqrtf(red[1] + LN_EPS) * g1[t] + be1[t];
  buf[t] = a; __syncthreads();

  float h = b2[t];
  for (int i = 0; i < 128; ++i) h = fmaf(buf[i], W2[i * 128 + t], h);
  h = fmaxf(h, 0.f);
  __syncthreads();
  buf[t] = h; __syncthreads();
  if (t == 0) { float s = 0; for (int i = 0; i < 128; ++i) s += buf[i]; red[0] = s * (1.f/128.f); }
  __syncthreads();
  d = h - red[0];
  buf[t] = d * d; __syncthreads();
  if (t == 0) { float s = 0; for (int i = 0; i < 128; ++i) s += buf[i]; red[1] = s * (1.f/128.f); }
  __syncthreads();
  h = d * rsqrtf(red[1] + LN_EPS) * g2[t] + be2[t];
  buf[t] = h * W3[t]; __syncthreads();
  if (t == 0) { float s = 0; for (int i = 0; i < 128; ++i) s += buf[i]; out[0] = s + b3[0]; }
}

__global__ void zero_ws_kernel(float* __restrict__ p, long n) {
  long i = (long)blockIdx.x * blockDim.x + threadIdx.x;
  if (i < n) p[i] = 0.0f;
}

// ---------------------------------------------------------------------------
extern "C" void kernel_launch(void* const* d_in, const int* in_sizes, int n_in,
                              void* d_out, int out_size, void* d_ws,
                              size_t ws_size, hipStream_t stream) {
  (void)n_in; (void)out_size; (void)ws_size;
  const float* x     = (const float*)d_in[0];
  const int*   eidx  = (const int*)d_in[1];   // [2,E] int32
  const float* eattr = (const float*)d_in[2];
  const float* u     = (const float*)d_in[3];
  // d_in[4] = batch (unused)
  const int N = in_sizes[0] / 128;  // 40000
  const int E = in_sizes[2] / 128;  // 640000
  const int* esrc = eidx;
  const int* edst = eidx + E;

  const float *ep[10], *np[10], *gp[10];
  for (int i = 0; i < 10; ++i) {
    ep[i] = (const float*)d_in[5 + i];
    np[i] = (const float*)d_in[15 + i];
    gp[i] = (const float*)d_in[25 + i];
  }

  float* x_new = (float*)d_out;
  float* e_new = x_new + (size_t)N * 128;
  float* u_new = e_new + (size_t)E * 128;

  // Workspace: agg[N,128] | edge_sum[128] | node_sum[128]  (~20.5 MB)
  float* agg      = (float*)d_ws;
  float* edge_sum = agg + (size_t)N * 128;
  float* node_sum = edge_sum + 128;

  hipFuncSetAttribute(reinterpret_cast<const void*>(edge_mlp_kernel),
                      hipFuncAttributeMaxDynamicSharedMemorySize, 131072);
  hipFuncSetAttribute(reinterpret_cast<const void*>(node_mlp_kernel),
                      hipFuncAttributeMaxDynamicSharedMemorySize, 131072);

  const long zn = (long)N * 128 + 256;
  zero_ws_kernel<<<(int)((zn + 255) / 256), 256, 0, stream>>>((float*)d_ws, zn);

  edge_mlp_kernel<<<E / 128, 256, 131072, stream>>>(
      x, esrc, edst, eattr, u,
      ep[0], ep[1], ep[2], ep[3], ep[4], ep[5], ep[6], ep[7], ep[8], ep[9],
      e_new, agg, edge_sum);

  node_mlp_kernel<<<(N + 127) / 128, 256, 131072, stream>>>(
      x, agg, u,
      np[0], np[1], np[2], np[3], np[4], np[5], np[6], np[7], np[8], np[9],
      x_new, node_sum, N);

  glob_mlp_kernel<<<1, 128, 0, stream>>>(
      u, node_sum, edge_sum,
      gp[0], gp[1], gp[2], gp[3], gp[4], gp[5], gp[6], gp[7], gp[8], gp[9],
      u_new);
}
